// MoEGate_2937757630475
// MI455X (gfx1250) — compile-verified, matrix operand order checked by reading, not run on hardware
//
#include <hip/hip_runtime.h>
#include <hip/hip_bf16.h>

typedef __attribute__((ext_vector_type(16))) _Float16 v16h;
typedef __attribute__((ext_vector_type(8)))  _Float16 v8h;
typedef __attribute__((ext_vector_type(4)))  _Float16 v4h;
typedef __attribute__((ext_vector_type(8)))  float    v8f;

#define HDIM   7168
#define NEXP   256
#define NGROUP 8
#define EPG    32        // experts per group
#define TOPKG  4
#define TOPK   8
#define RSCALE 2.5f
#define BM     32        // tokens per block (2 WMMA m-tiles)
#define BK     64        // K-tile per iteration (2 wmma k-steps)
#define NIT    (HDIM / BK)
#define LDA    72        // padded LDS row stride (f16 elements)

// ---------------------------------------------------------------------------
// One-shot conversion of the router weight [E, H] from f32 to f16 into d_ws.
// ---------------------------------------------------------------------------
__global__ __launch_bounds__(256) void wcvt_kernel(const float* __restrict__ W,
                                                   _Float16* __restrict__ Wh,
                                                   int n4) {
  int i = blockIdx.x * blockDim.x + threadIdx.x;
  if (i < n4) {
    const float4 w = ((const float4*)W)[i];
    v4h h = { (_Float16)w.x, (_Float16)w.y, (_Float16)w.z, (_Float16)w.w };
    ((v4h*)Wh)[i] = h;
  }
}

// ---------------------------------------------------------------------------
// Fused router kernel: 32 tokens per block, 8 waves. Each wave: 32 experts
// (two n-tiles) x 32 tokens (two m-tiles) -> 4 accumulators held in place.
// A tile double-buffered in LDS via pointer swap (one barrier per K-tile,
// single loop schedule -> no accumulator v_mov churn). B fragments stream
// straight from the L2-resident f16 weight copy. Epilogue fused in LDS.
// ---------------------------------------------------------------------------
__global__ __launch_bounds__(256) void moe_gate_kernel(
    const float* __restrict__ x,       // [N, H] f32
    const _Float16* __restrict__ Wh,   // [E, H] f16 (converted)
    const float* __restrict__ bias,    // [E]
    int* __restrict__ idx_out,         // [N, TOPK]
    float* __restrict__ w_out,         // [N, TOPK]
    int ntok) {
  __shared__ _Float16 As[2][BM * LDA];   // double-buffered A tile (f16)
  __shared__ float sfc[BM * NEXP];       // sigmoid(logit) + bias
  __shared__ float gsc[BM * NGROUP];     // group scores (top-2 sums)

  const int tid  = threadIdx.x;
  const int wid  = tid >> 5;
  const int lane = tid & 31;
  const int m0   = blockIdx.x * BM;

  const int eb0 = wid * 32;              // this wave's expert tiles
  const int eb1 = eb0 + 16;

  // WMMA 16-bit fragment mapping (ISA 7.12.2):
  // lane<16 : row/col = lane,   K-chunk bases {0,16}
  // lane>=16: row/col = lane-16, K-chunk bases {8,24}
  const int half = lane >> 4;
  const int frow = lane & 15;

  // cooperative A staging: thread -> (row, 8 consecutive k)
  const int srow = tid >> 3;             // 0..31
  const int scol = (tid & 7) << 3;       // 0..56
  const int sdst = srow * LDA + scol;

  // per-thread addresses for both LDS buffers (pointer-swap double buffer)
  _Float16* st[2] = { &As[0][sdst], &As[1][sdst] };
  const int foff = frow * LDA + half * 8;
  const _Float16* fr[2] = { &As[0][foff], &As[1][foff] };

  const float* xs = x + (size_t)(m0 + srow) * HDIM + scol;
  const _Float16* wq0 = Wh + (size_t)(eb0 + frow) * HDIM + half * 8;
  const _Float16* wq1 = Wh + (size_t)(eb1 + frow) * HDIM + half * 8;

  v8f acc00 = {}, acc01 = {}, acc10 = {}, acc11 = {};

  // one K-tile of WMMAs from read buffer `frd` and current weight pointers
  auto compute_tile = [&](const _Float16* frd, const _Float16* w0,
                          const _Float16* w1) {
#pragma unroll
    for (int ks = 0; ks < BK; ks += 32) {
      v8h b0l = *(const v8h*)(w0 + ks);
      v8h b0h = *(const v8h*)(w0 + ks + 16);
      v8h b1l = *(const v8h*)(w1 + ks);
      v8h b1h = *(const v8h*)(w1 + ks + 16);
      v16h bf0 = __builtin_shufflevector(b0l, b0h,
          0,1,2,3,4,5,6,7,8,9,10,11,12,13,14,15);
      v16h bf1 = __builtin_shufflevector(b1l, b1h,
          0,1,2,3,4,5,6,7,8,9,10,11,12,13,14,15);

      v8h a0l = *(const v8h*)(frd + ks);
      v8h a0h = *(const v8h*)(frd + ks + 16);
      v16h af0 = __builtin_shufflevector(a0l, a0h,
          0,1,2,3,4,5,6,7,8,9,10,11,12,13,14,15);
      acc00 = __builtin_amdgcn_wmma_f32_16x16x32_f16(
          false, af0, false, bf0, (short)0, acc00, false, false);
      acc01 = __builtin_amdgcn_wmma_f32_16x16x32_f16(
          false, af0, false, bf1, (short)0, acc01, false, false);

      v8h a1l = *(const v8h*)(frd + 16 * LDA + ks);
      v8h a1h = *(const v8h*)(frd + 16 * LDA + ks + 16);
      v16h af1 = __builtin_shufflevector(a1l, a1h,
          0,1,2,3,4,5,6,7,8,9,10,11,12,13,14,15);
      acc10 = __builtin_amdgcn_wmma_f32_16x16x32_f16(
          false, af1, false, bf0, (short)0, acc10, false, false);
      acc11 = __builtin_amdgcn_wmma_f32_16x16x32_f16(
          false, af1, false, bf1, (short)0, acc11, false, false);
    }
  };

  // prologue: stage K-tile 0 into buffer 0
  {
    float4 a0 = *(const float4*)(xs);
    float4 a1 = *(const float4*)(xs + 4);
    v8h h = { (_Float16)a0.x, (_Float16)a0.y, (_Float16)a0.z, (_Float16)a0.w,
              (_Float16)a1.x, (_Float16)a1.y, (_Float16)a1.z, (_Float16)a1.w };
    *(v8h*)st[0] = h;
  }
  __syncthreads();

  xs += BK;
  _Float16*       stw = st[1];           // staging target
  const _Float16* frd = fr[0];           // compute source
  _Float16*       sto = st[0];           // staging target after swap

#pragma clang loop unroll(disable)
  for (int it = 0; it < NIT - 1; ++it) {
    // stage next K-tile (overlaps with the WMMAs below)
    float4 a0 = *(const float4*)(xs);
    float4 a1 = *(const float4*)(xs + 4);
    __builtin_prefetch(xs + 2 * BK, 0, 1);            // global_prefetch_b8
    __builtin_prefetch(wq0 + 2 * BK, 0, 1);
    __builtin_prefetch(wq1 + 2 * BK, 0, 1);
    v8h h = { (_Float16)a0.x, (_Float16)a0.y, (_Float16)a0.z, (_Float16)a0.w,
              (_Float16)a1.x, (_Float16)a1.y, (_Float16)a1.z, (_Float16)a1.w };
    *(v8h*)stw = h;
    xs += BK;

    compute_tile(frd, wq0, wq1);
    wq0 += BK;
    wq1 += BK;

    // swap read/write buffers (no parity in addressing -> single schedule)
    const _Float16* tf = frd;
    frd = stw;
    stw = sto;
    sto = (_Float16*)stw;                 // keep third name consistent
    sto = (_Float16*)tf;
    __syncthreads();
  }

  // final (peeled) K-tile
  compute_tile(frd, wq0, wq1);

  // ---- scores -> LDS: sigmoid(logit) + bias ----
  // C/D layout: VGPR r, lane<16 -> (M=r, N=lane); lane>=16 -> (M=r+8, N=lane-16)
  {
    const int e0 = eb0 + frow;
    const int e1 = eb1 + frow;
    const float bb0 = bias[e0];
    const float bb1 = bias[e1];
#pragma unroll
    for (int r = 0; r < 8; ++r) {
      const int t0 = half * 8 + r;        // m-tile 0
      const int t1 = 16 + t0;             // m-tile 1
      sfc[t0 * NEXP + e0] = 1.f / (1.f + __expf(-acc00[r])) + bb0;
      sfc[t0 * NEXP + e1] = 1.f / (1.f + __expf(-acc01[r])) + bb1;
      sfc[t1 * NEXP + e0] = 1.f / (1.f + __expf(-acc10[r])) + bb0;
      sfc[t1 * NEXP + e1] = 1.f / (1.f + __expf(-acc11[r])) + bb1;
    }
  }
  __syncthreads();

  // ---- group scores: exactly one (token, group) pair per thread ----
  {
    const int t = tid >> 3, g = tid & 7;   // BM*NGROUP == 256
    const float* pq = sfc + t * NEXP + g * EPG;
    float m1 = -__builtin_inff(), m2 = -__builtin_inff();
    for (int i = 0; i < EPG; ++i) {
      float v = pq[i];
      if (v > m1) { m2 = m1; m1 = v; }
      else if (v > m2) { m2 = v; }
    }
    gsc[t * NGROUP + g] = m1 + m2;
  }
  __syncthreads();

  // ---- per-token: top-4 groups, masked top-8 experts, normalize, scale ----
  if (tid < BM && (m0 + tid) < ntok) {
    const int t = tid;
    float gs[NGROUP];
#pragma unroll
    for (int g = 0; g < NGROUP; ++g) gs[g] = gsc[t * NGROUP + g];

    unsigned gmask = 0;
    for (int j = 0; j < TOPKG; ++j) {      // ties -> lowest index
      int best = 0; float bv = -__builtin_inff();
      for (int g = 0; g < NGROUP; ++g)
        if (!((gmask >> g) & 1u) && gs[g] > bv) { bv = gs[g]; best = g; }
      gmask |= 1u << best;
    }

    float* p = sfc + t * NEXP;
    int wi[TOPK]; float wv[TOPK]; float wsum = 0.f;
    for (int j = 0; j < TOPK; ++j) {       // 8 masked argmax passes
      float bv = -__builtin_inff(); int bi = 0;
      for (int g = 0; g < NGROUP; ++g) {
        if (!((gmask >> g) & 1u)) continue;
        const float* q = p + g * EPG;
        for (int i = 0; i < EPG; ++i) {
          float v = q[i];
          if (v > bv) { bv = v; bi = g * EPG + i; }
        }
      }
      wi[j] = bi;
      float sc = bv - bias[bi];            // recover raw sigmoid score
      wv[j] = sc; wsum += sc;
      p[bi] = -__builtin_inff();           // knock out for next pass
    }

    const float inv = RSCALE / (wsum + 1e-20f);
    const int n = m0 + t;
#pragma unroll
    for (int j = 0; j < TOPK; ++j) {
      idx_out[n * TOPK + j] = wi[j];
      w_out[n * TOPK + j]   = wv[j] * inv;
    }
  }
}

// ---------------------------------------------------------------------------
extern "C" void kernel_launch(void* const* d_in, const int* in_sizes, int n_in,
                              void* d_out, int out_size, void* d_ws, size_t ws_size,
                              hipStream_t stream) {
  const float* x    = (const float*)d_in[0];   // [B,S,H] f32
  const float* W    = (const float*)d_in[1];   // [E,H]   f32
  const float* bias = (const float*)d_in[2];   // [E]     f32

  const int ntok = in_sizes[0] / HDIM;         // B*S

  _Float16* Wh = (_Float16*)d_ws;              // f16 weight copy (E*H*2 bytes)
  const int n4 = (NEXP * HDIM) / 4;
  wcvt_kernel<<<(n4 + 255) / 256, 256, 0, stream>>>(W, Wh, n4);

  int*   idx_out = (int*)d_out;                          // first N*K ints
  float* w_out   = (float*)d_out + (size_t)ntok * TOPK;  // next N*K floats
  moe_gate_kernel<<<(ntok + BM - 1) / BM, 256, 0, stream>>>(x, Wh, bias,
                                                            idx_out, w_out, ntok);
}
